// VWE_87651692577327
// MI455X (gfx1250) — matrix operand order repaired and checked by reading, not compiled, not run on hardware
//
#include <hip/hip_runtime.h>

typedef _Float16 half_t;
typedef _Float16 v16h __attribute__((ext_vector_type(16)));
typedef _Float16 h8   __attribute__((ext_vector_type(8)));
typedef float    v8f  __attribute__((ext_vector_type(8)));
typedef float    f2v  __attribute__((ext_vector_type(2)));
typedef unsigned int u32;

#define BATCH 8
#define CH    2048
#define HWSZ  4096
#define KW    256
#define PADW  20            // u32 pitch per column in LDS staging (conflict-free, 16B aligned)

// output layout (floats): x_corr | x_hist | y_word | new_centroid
#define HIST_OFF  8388608
#define YW_OFF    8390656
#define CTR_OFF   8392704

// -------------------------------------------------------------------------
// K0: zero y_word, ctr accumulator (lives in d_out), counts
__global__ void k_init(float* __restrict__ out, u32* __restrict__ counts) {
  int i = blockIdx.x * 256 + threadIdx.x;          // 0..524287
  out[CTR_OFF + i] = 0.0f;
  if (i < 2048) out[YW_OFF + i] = 0.0f;
  if (i < 256)  counts[i] = 0u;
}

// -------------------------------------------------------------------------
// K1: L2-normalize codebook rows, store f16 copy
__global__ void k_wnorm(const float* __restrict__ centroid, half_t* __restrict__ w16) {
  __shared__ float red[256];
  int k = blockIdx.x, t = threadIdx.x;
  const float* row = centroid + k * CH;
  float s = 0.f;
  for (int c = t; c < CH; c += 256) { float v = row[c]; s += v * v; }
  red[t] = s; __syncthreads();
  for (int st = 128; st > 0; st >>= 1) { if (t < st) red[t] += red[t + st]; __syncthreads(); }
  float rinv = 1.0f / fmaxf(sqrtf(red[0]), 1e-12f);
  for (int c = t; c < CH; c += 256) w16[k * CH + c] = (half_t)(row[c] * rinv);
}

// -------------------------------------------------------------------------
// K2: fused  W_norm[256x2048] x X[2048x64cols] WMMA GEMM  + column-norm +
//     softmax over K + argmax/labels + y_word + counts + hist partials.
// grid: 512 blocks (8 b x 64 col-blocks of 64), 512 threads (16 waves).
__global__ __launch_bounds__(512) void k_logits(
    const float* __restrict__ x, const half_t* __restrict__ w16,
    float* __restrict__ xcorr, int* __restrict__ labels, float* __restrict__ yword,
    u32* __restrict__ counts, float* __restrict__ histpart) {
  __shared__ __align__(16) u32 sx[2][64 * PADW];   // staged f16 x, pair-packed along c
  __shared__ float slog[KW * 64];                  // logits -> probs (64 KB)
  __shared__ float ssq[64];
  __shared__ float rmax[512]; __shared__ int ridx[512]; __shared__ float rsum[512];

  const int t = threadIdx.x;
  const int bx = blockIdx.x;
  const int b = bx >> 6;
  const int hwbase = (bx & 63) << 6;               // 64 columns per block
  const float* xb = x + (size_t)b * CH * HWSZ;

  // staging: thread covers 2 adjacent columns, one c-pair; f2v (b64) NT loads
  const int colpair = t & 31;                      // -> cols 2*colpair, 2*colpair+1
  const int cpair = t >> 5;                        // 0..15
  float ss0 = 0.f, ss1 = 0.f;
  if (t < 64) ssq[t] = 0.f;

  auto stage = [&](int i) {
    u32* buf = sx[i & 1];
    int c = (i << 5) + cpair * 2;
    const f2v* p0 = (const f2v*)&xb[(size_t)c * HWSZ + hwbase + colpair * 2];
    const f2v* p1 = (const f2v*)&xb[(size_t)(c + 1) * HWSZ + hwbase + colpair * 2];
    f2v a = __builtin_nontemporal_load(p0);        // row c   : cols (x), (y)
    f2v bb = __builtin_nontemporal_load(p1);       // row c+1
    ss0 += a.x * a.x + bb.x * bb.x;
    ss1 += a.y * a.y + bb.y * bb.y;
    union { half_t h[2]; u32 u; } P0, P1;
    P0.h[0] = (half_t)a.x; P0.h[1] = (half_t)bb.x;
    P1.h[0] = (half_t)a.y; P1.h[1] = (half_t)bb.y;
    buf[(colpair * 2) * PADW + cpair]     = P0.u;
    buf[(colpair * 2 + 1) * PADW + cpair] = P1.u;
  };

  const int wave = t >> 5, lane = t & 31;
  const int lhalf = lane >> 4, ln = lane & 15;

  v8f zero = {};
  v8f acc[4]; acc[0] = zero; acc[1] = zero; acc[2] = zero; acc[3] = zero;

  stage(0);
  for (int i = 0; i < 64; ++i) {
    __syncthreads();
    if (i + 1 < 64) stage(i + 1);
    const u32* buf = sx[i & 1];
    int c0 = i << 5;
    // A fragment: w_norm rows, documented f16 16x32 layout
    const half_t* wp = w16 + (size_t)((wave << 4) + ln) * CH + c0 + (lhalf ? 8 : 0);
    union { h8 h[2]; v16h v; } A;
    A.h[0] = *(const h8*)wp;
    A.h[1] = *(const h8*)(wp + 16);
#pragma unroll
    for (int j = 0; j < 4; ++j) {
      const u32* bp = &buf[(16 * j + ln) * PADW + (lhalf ? 4 : 0)];
      union { uint4 q[2]; v16h v; } Bf;
      Bf.q[0] = *(const uint4*)(bp);
      Bf.q[1] = *(const uint4*)(bp + 8);
      acc[j] = __builtin_amdgcn_wmma_f32_16x16x32_f16(false, A.v, false, Bf.v,
                                                      (short)0, acc[j], false, false);
    }
  }
  atomicAdd(&ssq[colpair * 2], ss0);               // finish column sum-of-squares
  atomicAdd(&ssq[colpair * 2 + 1], ss1);
  __syncthreads();

  // scale by 1/||x_col|| and park logits in LDS
#pragma unroll
  for (int j = 0; j < 4; ++j) {
    int cc = 16 * j + ln;
    float rn = 1.0f / fmaxf(sqrtf(ssq[cc]), 1e-12f);
#pragma unroll
    for (int r = 0; r < 8; ++r) {
      int k = (wave << 4) + (lhalf << 3) + r;      // C/D layout: lanes16-31 -> M+8
      slog[k * 64 + cc] = acc[j][r] * rn;
    }
  }
  __syncthreads();

  // softmax / argmax: 8 segments of 32 k per column
  {
    int c2 = t & 63, seg = t >> 6, kb = seg << 5;
    float m = -3.4e38f; int mi = 0;
#pragma unroll 4
    for (int q = 0; q < 32; ++q) { float v = slog[(kb + q) * 64 + c2]; if (v > m) { m = v; mi = kb + q; } }
    rmax[c2 * 8 + seg] = m; ridx[c2 * 8 + seg] = mi;
    __syncthreads();
    float cm = -3.4e38f; int ci = 0;
    for (int sg = 0; sg < 8; ++sg) {               // ascending -> first-index tiebreak
      float v = rmax[c2 * 8 + sg];
      if (v > cm) { cm = v; ci = ridx[c2 * 8 + sg]; }
    }
    float se = 0.f;
#pragma unroll 4
    for (int q = 0; q < 32; ++q) {
      float e = __expf(slog[(kb + q) * 64 + c2] - cm);
      se += e; slog[(kb + q) * 64 + c2] = e;
    }
    rsum[c2 * 8 + seg] = se;
    __syncthreads();
    float tot = 0.f;
    for (int sg = 0; sg < 8; ++sg) tot += rsum[c2 * 8 + sg];
    float inv = 1.0f / tot;
    size_t obase = ((size_t)b * KW + kb) * HWSZ + hwbase + c2;
#pragma unroll 4
    for (int q = 0; q < 32; ++q) {
      float p = slog[(kb + q) * 64 + c2] * inv;
      slog[(kb + q) * 64 + c2] = p;
      __builtin_nontemporal_store(p, &xcorr[obase + (size_t)q * HWSZ]);  // write-once
    }
    if (seg == 0) {
      labels[b * HWSZ + hwbase + c2] = ci;
      yword[b * KW + ci] = 1.0f;                   // benign same-value race
      atomicAdd(&counts[ci], 1u);                  // integer atomics: exact
    }
    __syncthreads();
    if (t < KW) {                                  // per-block hist partial
      float s = 0.f;
      for (int cc = 0; cc < 64; ++cc) s += slog[t * 64 + cc];
      histpart[(size_t)bx * KW + t] = s;
    }
  }
}

// -------------------------------------------------------------------------
// K3: ctr = Onehot[256 x 4096] x X^T[4096 x 64c] per batch via WMMA,
//     atomic-combined over batches.  grid: 256 (8 b x 32 c-blocks), 512 thr.
__global__ __launch_bounds__(512) void k_ctr(
    const float* __restrict__ x, const int* __restrict__ labels, float* __restrict__ ctr) {
  __shared__ __align__(16) u32 sb[2][64 * PADW];   // x packed along hw pairs, per c column
  __shared__ int slab[2][32];
  const int t = threadIdx.x;
  const int bx = blockIdx.x;
  const int b = bx >> 5;
  const int c0 = (bx & 31) << 6;                   // 64 c-columns
  const float* xb = x + (size_t)b * CH * HWSZ;
  const int* labb = labels + b * HWSZ;

  const int hp = t & 15;                           // hw pair
  const int cr = t >> 4;                           // 0..31

  auto stage = [&](int i) {
    int hw0 = i << 5;
    u32* buf = sb[i & 1];
#pragma unroll
    for (int q = 0; q < 2; ++q) {
      int crow = cr + q * 32;
      const f2v* p = (const f2v*)&xb[(size_t)(c0 + crow) * HWSZ + hw0 + hp * 2];
      f2v v = __builtin_nontemporal_load(p);       // contiguous hw pair
      union { half_t h[2]; u32 u; } P;
      P.h[0] = (half_t)v.x; P.h[1] = (half_t)v.y;
      buf[crow * PADW + hp] = P.u;
    }
    if (t < 32) slab[i & 1][t] = labb[hw0 + t];
  };

  const int wave = t >> 5, lane = t & 31;
  const int lhalf = lane >> 4, ln = lane & 15;
  const int myk = (wave << 4) + ln;

  v8f zero = {};
  v8f acc[4]; acc[0] = zero; acc[1] = zero; acc[2] = zero; acc[3] = zero;

  stage(0);
  for (int i = 0; i < 128; ++i) {
    __syncthreads();
    if (i + 1 < 128) stage(i + 1);
    const u32* buf = sb[i & 1];
    const int* lab = slab[i & 1];
    union { half_t h[16]; v16h v; } A;             // one-hot rows, A 16x32 layout
#pragma unroll
    for (int q = 0; q < 8; ++q) {
      A.h[q]     = (lab[(lhalf ? 8 : 0) + q]      == myk) ? (half_t)1.0f : (half_t)0.0f;
      A.h[8 + q] = (lab[16 + (lhalf ? 8 : 0) + q] == myk) ? (half_t)1.0f : (half_t)0.0f;
    }
#pragma unroll
    for (int j = 0; j < 4; ++j) {
      const u32* bp = &buf[(16 * j + ln) * PADW + (lhalf ? 4 : 0)];
      union { uint4 q[2]; v16h v; } Bf;
      Bf.q[0] = *(const uint4*)(bp);
      Bf.q[1] = *(const uint4*)(bp + 8);
      acc[j] = __builtin_amdgcn_wmma_f32_16x16x32_f16(false, A.v, false, Bf.v,
                                                      (short)0, acc[j], false, false);
    }
  }
#pragma unroll
  for (int j = 0; j < 4; ++j) {
    int cc = c0 + 16 * j + ln;
#pragma unroll
    for (int r = 0; r < 8; ++r) {
      int k = (wave << 4) + (lhalf << 3) + r;
      atomicAdd(&ctr[(size_t)k * CH + cc], acc[j][r]);
    }
  }
}

// -------------------------------------------------------------------------
// K4: new_centroid = rho*ctr/(count+eps) + (1-rho)*centroid ; reduce hist partials
__global__ void k_fin(const float* __restrict__ centroid, const u32* __restrict__ counts,
                      const float* __restrict__ histpart, float* __restrict__ out) {
  int i = blockIdx.x * 256 + threadIdx.x;          // 0..524287
  float v = out[CTR_OFF + i];
  float cnt = (float)counts[i >> 11];
  out[CTR_OFF + i] = 0.01f * (v / (cnt + 1e-4f)) + 0.99f * centroid[i];
  if (i < 2048) {
    int b = i >> 8, k = i & 255;
    float s = 0.f;
    for (int cb = 0; cb < 64; ++cb) s += histpart[(size_t)(b * 64 + cb) * KW + k];
    out[HIST_OFF + i] = s;
  }
}

// -------------------------------------------------------------------------
extern "C" void kernel_launch(void* const* d_in, const int* in_sizes, int n_in,
                              void* d_out, int out_size, void* d_ws, size_t ws_size,
                              hipStream_t stream) {
  const float* x        = (const float*)d_in[0];
  const float* centroid = (const float*)d_in[1];
  float* out = (float*)d_out;

  char* ws = (char*)d_ws;
  half_t* w16     = (half_t*)ws;                           // 1 MB
  int*    labels  = (int*)(ws + (1u << 20));               // 128 KB
  u32*    counts  = (u32*)(ws + (1u << 20) + 131072);      // 1 KB
  float*  histpart= (float*)(ws + (1u << 20) + 131072 + 1024); // 512 KB

  k_init  <<<2048, 256, 0, stream>>>(out, counts);
  k_wnorm <<<KW,   256, 0, stream>>>(centroid, w16);
  k_logits<<<512,  512, 0, stream>>>(x, w16, out, labels, out + YW_OFF, counts, histpart);
  k_ctr   <<<256,  512, 0, stream>>>(x, labels, out + CTR_OFF);
  k_fin   <<<2048, 256, 0, stream>>>(centroid, counts, histpart, out);
}
